// MoEHybridTransformerBlockBase_47150150975998
// MI455X (gfx1250) — matrix operand order repaired and checked by reading, not compiled, not run on hardware
//
#include <hip/hip_runtime.h>
#include <hip/hip_bf16.h>
#include <cstddef>
#include <cstdint>

// ---------------------------------------------------------------------------
// MoE hybrid transformer block for MI455X (gfx1250, wave32, WMMA).
// - bf16 WMMA (v_wmma_f32_16x16x32_bf16) for every GEMM, fp32 accumulate.
// - GEMM global->LDS staging uses CDNA5 GLOBAL_LOAD_ASYNC_TO_LDS_B128
//   (ASYNCcnt) with double-buffered LDS so the DMA overlaps the WMMA pipe.
// ---------------------------------------------------------------------------

typedef __bf16 bf16;
typedef __attribute__((ext_vector_type(8)))  __bf16 bf16x8;
typedef __attribute__((ext_vector_type(16))) __bf16 v16bf;
typedef __attribute__((ext_vector_type(8)))  float  v8f;

static constexpr int Bd  = 2;
static constexpr int Sd  = 2048;
static constexpr int Dd  = 1024;
static constexpr int Hh  = 16;
static constexpr int HDd = 64;
static constexpr int FFd = 4096;
static constexpr int Ed  = 8;
static constexpr int EHd = 1024;
static constexpr int Td  = Bd * Sd;   // 4096 tokens

// ---------------- fragment helpers (ISA 7.12.2 layouts) --------------------
// A 16x32 bf16 frag: lane l -> row l&15, K = kb..kb+7 (elems 0..7) and
// K = kb+16..kb+23 (elems 8..15), kb = (l<16 ? 0 : 8).
// B 32x16 bf16 frag: lane l -> col l&15, same K split as A (kb..kb+7, kb+16..kb+23).
__device__ __forceinline__ v16bf make_frag(const bf16* p0, const bf16* p1) {
  bf16x8 lo = *(const bf16x8*)p0;
  bf16x8 hi = *(const bf16x8*)p1;
  v16bf r;
#pragma unroll
  for (int i = 0; i < 8; ++i) { r[i] = lo[i]; r[8 + i] = hi[i]; }
  return r;
}

// CDNA5 async global->LDS copy, 16B per lane, tracked by ASYNCcnt.
// lds_off: LDS byte address (low 32 bits of a generic __shared__ pointer).
__device__ __forceinline__ void async_copy_b128(unsigned lds_off,
                                                const bf16* gptr) {
  asm volatile("global_load_async_to_lds_b128 %0, %1, off"
               :: "v"(lds_off), "v"((unsigned long long)(uintptr_t)gptr)
               : "memory");
}
__device__ __forceinline__ void wait_async0() {
  asm volatile("s_wait_asynccnt 0x0" ::: "memory");
}

__device__ __forceinline__ float wave_half_max(float v) {
#pragma unroll
  for (int m = 8; m >= 1; m >>= 1) v = fmaxf(v, __shfl_xor(v, m, 32));
  return v;
}
__device__ __forceinline__ float wave_half_sum(float v) {
#pragma unroll
  for (int m = 8; m >= 1; m >>= 1) v += __shfl_xor(v, m, 32);
  return v;
}
__device__ __forceinline__ float wave_sum32(float v) {
#pragma unroll
  for (int m = 16; m >= 1; m >>= 1) v += __shfl_xor(v, m, 32);
  return v;
}

// ---------------------------- elementwise casts ----------------------------
__global__ __launch_bounds__(256) void cast_bf16_kernel(
    const float* __restrict__ in, bf16* __restrict__ out, size_t n) {
  size_t stride = (size_t)gridDim.x * blockDim.x;
  for (size_t i = (size_t)blockIdx.x * blockDim.x + threadIdx.x; i < n; i += stride)
    out[i] = (bf16)in[i];
}

// W[K][N] fp32 -> Wt[N][K] bf16 (B^T layout for WMMA B-fragment loads)
__global__ __launch_bounds__(256) void tcast_bf16_kernel(
    const float* __restrict__ W, bf16* __restrict__ Wt, int K, int N) {
  size_t total = (size_t)K * N;
  size_t stride = (size_t)gridDim.x * blockDim.x;
  for (size_t i = (size_t)blockIdx.x * blockDim.x + threadIdx.x; i < total; i += stride) {
    int n = (int)(i / K);
    int k = (int)(i % K);
    Wt[i] = (bf16)W[(size_t)k * N + n];
  }
}

// ------------------------------- WMMA GEMM ---------------------------------
// C[M][N] (fp32 acc) = A[M][K] (bf16, row major) @ Bt[N][K]^T (bf16)
// Block tile 128x128, 8 waves, each wave 64(M)x32(N) = 4x2 wmma tiles.
// Double-buffered LDS filled by GLOBAL_LOAD_ASYNC_TO_LDS_B128.
#define GBM 128
#define GBN 128
#define GBK 32
#define GKP 40   // LDS row stride (elems): 80B rows, 16B aligned, de-conflicted

__global__ __launch_bounds__(256) void gemm_bf16_kernel(
    const bf16* __restrict__ A, const bf16* __restrict__ Bt,
    int M, int N, int K,
    float* __restrict__ Cf, int accC,
    const float* __restrict__ rowScale, int rsStride,
    bf16* __restrict__ Cb, bf16* __restrict__ CbT, int ldt,
    const float* __restrict__ partner) {
  __shared__ __align__(16) bf16 As[2][GBM * GKP];
  __shared__ __align__(16) bf16 Bs[2][GBM * GKP];

  const int tid  = threadIdx.x;
  const int wave = tid >> 5;
  const int lane = tid & 31;
  const int half = lane >> 4;
  const int nl   = lane & 15;
  const int m0 = blockIdx.y * GBM;
  const int n0 = blockIdx.x * GBN;
  const int wm = (wave >> 2) * 64;   // 0 / 64
  const int wn = (wave & 3) * 32;    // 0 / 32 / 64 / 96

  // this thread's two 16B staging chunks per tile
  const int r0 = tid >> 2,         c0 = tid & 3;          // chunk 0
  const int r1 = (tid + 256) >> 2, c1 = (tid + 256) & 3;  // chunk 1
  const unsigned l0 = (unsigned)((r0 * GKP + c0 * 8) * 2);
  const unsigned l1 = (unsigned)((r1 * GKP + c1 * 8) * 2);

  v8f acc[4][2];
#pragma unroll
  for (int mi = 0; mi < 4; ++mi)
#pragma unroll
    for (int ni = 0; ni < 2; ++ni)
#pragma unroll
      for (int r = 0; r < 8; ++r) acc[mi][ni][r] = 0.f;

  auto issue_tile = [&](int buf, int k0) {
    unsigned aB = (unsigned)(uintptr_t)(void*)&As[buf][0];
    unsigned bB = (unsigned)(uintptr_t)(void*)&Bs[buf][0];
    async_copy_b128(aB + l0, &A[(size_t)(m0 + r0) * K + k0 + c0 * 8]);
    async_copy_b128(aB + l1, &A[(size_t)(m0 + r1) * K + k0 + c1 * 8]);
    async_copy_b128(bB + l0, &Bt[(size_t)(n0 + r0) * K + k0 + c0 * 8]);
    async_copy_b128(bB + l1, &Bt[(size_t)(n0 + r1) * K + k0 + c1 * 8]);
  };

  const int nk = K / GBK;
  issue_tile(0, 0);
  wait_async0();
  __syncthreads();

  for (int kt = 0; kt < nk; ++kt) {
    const int buf = kt & 1;
    if (kt + 1 < nk) issue_tile(buf ^ 1, (kt + 1) * GBK);  // overlap with WMMA

    v16bf bfrag[2];
#pragma unroll
    for (int ni = 0; ni < 2; ++ni) {
      const bf16* bp = &Bs[buf][(wn + ni * 16 + nl) * GKP + half * 8];
      bfrag[ni] = make_frag(bp, bp + 16);
    }
#pragma unroll
    for (int mi = 0; mi < 4; ++mi) {
      const bf16* ap = &As[buf][(wm + mi * 16 + nl) * GKP + half * 8];
      v16bf afrag = make_frag(ap, ap + 16);
#pragma unroll
      for (int ni = 0; ni < 2; ++ni) {
        acc[mi][ni] = __builtin_amdgcn_wmma_f32_16x16x32_bf16(
            false, afrag, false, bfrag[ni], (short)0, acc[mi][ni], false, false);
      }
    }
    wait_async0();       // next tile landed in LDS
    __syncthreads();     // all waves done reading current tile
  }

  // epilogue: C layout -> lane l: col nl, vgpr r: row r + half*8
#pragma unroll
  for (int mi = 0; mi < 4; ++mi)
#pragma unroll
    for (int ni = 0; ni < 2; ++ni)
#pragma unroll
      for (int r = 0; r < 8; ++r) {
        int mg = m0 + wm + mi * 16 + half * 8 + r;
        int ng = n0 + wn + ni * 16 + nl;
        float v = acc[mi][ni][r];
        if (partner) {                                     // SwiGLU fuse
          float p = partner[(size_t)mg * N + ng];
          v *= p / (1.f + __expf(-p));
        }
        if (rowScale) v *= rowScale[(size_t)mg * rsStride];
        if (Cf) {
          size_t o = (size_t)mg * N + ng;
          Cf[o] = accC ? (Cf[o] + v) : v;
        }
        if (Cb)  Cb[(size_t)mg * N + ng]   = (bf16)v;
        if (CbT) CbT[(size_t)ng * ldt + mg] = (bf16)v;    // transposed store
      }
}

// --------------------------- flash attention -------------------------------
// grid = (S/128, H, B); 8 waves, each wave owns 16 query rows.
__global__ __launch_bounds__(256) void attn_kernel(
    const bf16* __restrict__ qbf, const bf16* __restrict__ kbf,
    const bf16* __restrict__ vT,  bf16* __restrict__ obf) {
  __shared__ bf16 pLDS[8][16 * 64];

  const int wave = threadIdx.x >> 5;
  const int lane = threadIdx.x & 31;
  const int half = lane >> 4;
  const int nl   = lane & 15;
  const int q0 = blockIdx.x * 128 + wave * 16;   // query tile base (in S)
  const int h  = blockIdx.y;
  const int b  = blockIdx.z;
  const size_t tb = (size_t)b * Sd;

  // Q fragments: 16x64 = 2 K-steps
  v16bf qf[2];
#pragma unroll
  for (int ks = 0; ks < 2; ++ks) {
    const bf16* qp = &qbf[(tb + q0 + nl) * Dd + h * HDd + ks * 32 + half * 8];
    qf[ks] = make_frag(qp, qp + 16);
  }

  v8f Oacc[4];
  float rm[8], ls[8];
#pragma unroll
  for (int ni = 0; ni < 4; ++ni)
#pragma unroll
    for (int r = 0; r < 8; ++r) Oacc[ni][r] = 0.f;
#pragma unroll
  for (int r = 0; r < 8; ++r) { rm[r] = -1e30f; ls[r] = 0.f; }

  const int kend = q0 + 15;                       // causal: keys <= last q row
  for (int kc = 0; kc <= kend; kc += 64) {
    v8f sacc[4];
#pragma unroll
    for (int ni = 0; ni < 4; ++ni)
#pragma unroll
      for (int r = 0; r < 8; ++r) sacc[ni][r] = 0.f;

    // S = Q @ K^T  (B^T = K row-major, contiguous)
#pragma unroll
    for (int ks = 0; ks < 2; ++ks)
#pragma unroll
      for (int ni = 0; ni < 4; ++ni) {
        const bf16* kp =
            &kbf[(tb + kc + ni * 16 + nl) * Dd + h * HDd + ks * 32 + half * 8];
        v16bf kf = make_frag(kp, kp + 16);
        sacc[ni] = __builtin_amdgcn_wmma_f32_16x16x32_bf16(
            false, qf[ks], false, kf, (short)0, sacc[ni], false, false);
      }

    // scale + causal mask
#pragma unroll
    for (int ni = 0; ni < 4; ++ni)
#pragma unroll
      for (int r = 0; r < 8; ++r) {
        float s = sacc[ni][r] * 0.125f;            // 1/sqrt(64)
        int qg = q0 + half * 8 + r;
        int kg = kc + ni * 16 + nl;
        sacc[ni][r] = (kg <= qg) ? s : -1e30f;
      }

    // online softmax per accumulator row
#pragma unroll
    for (int r = 0; r < 8; ++r) {
      float mx = sacc[0][r];
#pragma unroll
      for (int ni = 1; ni < 4; ++ni) mx = fmaxf(mx, sacc[ni][r]);
      mx = wave_half_max(mx);
      float nm  = fmaxf(rm[r], mx);
      float esc = __expf(rm[r] - nm);
      rm[r] = nm;
      float rs = 0.f;
#pragma unroll
      for (int ni = 0; ni < 4; ++ni) {
        float p = __expf(sacc[ni][r] - nm);
        sacc[ni][r] = p;
        rs += p;
      }
      rs = wave_half_sum(rs);
      ls[r] = ls[r] * esc + rs;
#pragma unroll
      for (int ni = 0; ni < 4; ++ni) Oacc[ni][r] *= esc;
      int row = half * 8 + r;
#pragma unroll
      for (int ni = 0; ni < 4; ++ni)
        pLDS[wave][row * 64 + ni * 16 + nl] = (bf16)sacc[ni][r];
    }

    // O += P @ V   (B^T = V^T row-major, contiguous along tokens)
#pragma unroll
    for (int ks = 0; ks < 2; ++ks) {
      const bf16* pp = &pLDS[wave][nl * 64 + ks * 32 + half * 8];
      v16bf pf = make_frag(pp, pp + 16);
#pragma unroll
      for (int ni = 0; ni < 4; ++ni) {
        const bf16* vp = &vT[(size_t)(h * HDd + ni * 16 + nl) * Td + tb + kc +
                             ks * 32 + half * 8];
        v16bf vf = make_frag(vp, vp + 16);
        Oacc[ni] = __builtin_amdgcn_wmma_f32_16x16x32_bf16(
            false, pf, false, vf, (short)0, Oacc[ni], false, false);
      }
    }
  }

#pragma unroll
  for (int r = 0; r < 8; ++r) {
    float inv = 1.f / ls[r];
#pragma unroll
    for (int ni = 0; ni < 4; ++ni)
      obf[(tb + q0 + half * 8 + r) * Dd + h * HDd + ni * 16 + nl] =
          (bf16)(Oacc[ni][r] * inv);
  }
}

// ------------------------------- router ------------------------------------
// one wave per token: logits = x @ router_w, softmax over 8, top-2 gate table
__global__ __launch_bounds__(256) void router_kernel(
    const float* __restrict__ x, const float* __restrict__ rw,
    float* __restrict__ gate) {
  const int wave = threadIdx.x >> 5;
  const int lane = threadIdx.x & 31;
  const int t = blockIdx.x * 8 + wave;
  float acc[Ed];
#pragma unroll
  for (int e = 0; e < Ed; ++e) acc[e] = 0.f;
  for (int d = lane; d < Dd; d += 32) {
    float xv = x[(size_t)t * Dd + d];
#pragma unroll
    for (int e = 0; e < Ed; ++e) acc[e] += xv * rw[(size_t)d * Ed + e];
  }
#pragma unroll
  for (int e = 0; e < Ed; ++e) acc[e] = wave_sum32(acc[e]);
  float mx = acc[0];
#pragma unroll
  for (int e = 1; e < Ed; ++e) mx = fmaxf(mx, acc[e]);
  float p[Ed], den = 0.f;
#pragma unroll
  for (int e = 0; e < Ed; ++e) { p[e] = __expf(acc[e] - mx); den += p[e]; }
  float inv = 1.f / den;
#pragma unroll
  for (int e = 0; e < Ed; ++e) p[e] *= inv;
  int i0 = 0;
#pragma unroll
  for (int e = 1; e < Ed; ++e) if (p[e] > p[i0]) i0 = e;
  int i1 = (i0 == 0) ? 1 : 0;
#pragma unroll
  for (int e = 0; e < Ed; ++e) if (e != i0 && p[e] > p[i1]) i1 = e;
  if (lane < Ed)
    gate[(size_t)t * Ed + lane] =
        (lane == i0) ? p[i0] : ((lane == i1) ? p[i1] : 0.f);
}

// --------------------------- norm / residual -------------------------------
__global__ __launch_bounds__(256) void rmsnorm_add_kernel(
    const float* __restrict__ base, const float* __restrict__ src,
    const float* __restrict__ w, float* __restrict__ out,
    bf16* __restrict__ out_bf) {
  __shared__ float red[8];
  const int t = blockIdx.x;
  const float* s = src + (size_t)t * Dd;
  float ss = 0.f;
  for (int d = threadIdx.x; d < Dd; d += 256) { float v = s[d]; ss += v * v; }
  ss = wave_sum32(ss);
  if ((threadIdx.x & 31) == 0) red[threadIdx.x >> 5] = ss;
  __syncthreads();
  float tot = 0.f;
#pragma unroll
  for (int i = 0; i < 8; ++i) tot += red[i];
  float r = rsqrtf(tot / Dd + 1e-5f);
  for (int d = threadIdx.x; d < Dd; d += 256) {
    float v = base[(size_t)t * Dd + d] + s[d] * r * w[d];
    out[(size_t)t * Dd + d] = v;
    out_bf[(size_t)t * Dd + d] = (bf16)v;
  }
}

// out = h + rmsnorm(ffn)*ffw + rmsnorm(moe)*mow
__global__ __launch_bounds__(256) void final_kernel(
    const float* __restrict__ h, const float* __restrict__ ffn,
    const float* __restrict__ ffw, const float* __restrict__ moe,
    const float* __restrict__ mow, float* __restrict__ out) {
  __shared__ float red1[8], red2[8];
  const int t = blockIdx.x;
  const float* f = ffn + (size_t)t * Dd;
  const float* m = moe + (size_t)t * Dd;
  float s1 = 0.f, s2 = 0.f;
  for (int d = threadIdx.x; d < Dd; d += 256) {
    float a = f[d], b = m[d];
    s1 += a * a; s2 += b * b;
  }
  s1 = wave_sum32(s1); s2 = wave_sum32(s2);
  if ((threadIdx.x & 31) == 0) { red1[threadIdx.x >> 5] = s1; red2[threadIdx.x >> 5] = s2; }
  __syncthreads();
  float t1 = 0.f, t2 = 0.f;
#pragma unroll
  for (int i = 0; i < 8; ++i) { t1 += red1[i]; t2 += red2[i]; }
  float r1 = rsqrtf(t1 / Dd + 1e-5f);
  float r2 = rsqrtf(t2 / Dd + 1e-5f);
  for (int d = threadIdx.x; d < Dd; d += 256)
    out[(size_t)t * Dd + d] =
        h[(size_t)t * Dd + d] + f[d] * r1 * ffw[d] + m[d] * r2 * mow[d];
}

// ------------------------------- host side ---------------------------------
static void launch_gemm(const bf16* A, const bf16* Bt, int M, int N, int K,
                        float* Cf, int accC, const float* rs, int rss,
                        bf16* Cb, bf16* CbT, int ldt, const float* partner,
                        hipStream_t stream) {
  dim3 g(N / GBN, M / GBM);
  gemm_bf16_kernel<<<g, 256, 0, stream>>>(A, Bt, M, N, K, Cf, accC, rs, rss,
                                          Cb, CbT, ldt, partner);
}

extern "C" void kernel_launch(void* const* d_in, const int* in_sizes, int n_in,
                              void* d_out, int out_size, void* d_ws, size_t ws_size,
                              hipStream_t stream) {
  (void)in_sizes; (void)n_in; (void)out_size; (void)ws_size;
  const float* x    = (const float*)d_in[0];
  const float* wq   = (const float*)d_in[1];
  const float* wk   = (const float*)d_in[2];
  const float* wv   = (const float*)d_in[3];
  const float* wo   = (const float*)d_in[4];
  const float* w1   = (const float*)d_in[5];
  const float* w2   = (const float*)d_in[6];
  const float* w3   = (const float*)d_in[7];
  const float* anw  = (const float*)d_in[8];
  const float* ffw  = (const float*)d_in[9];
  const float* mow  = (const float*)d_in[10];
  const float* rw   = (const float*)d_in[11];
  const float* ew1  = (const float*)d_in[12];
  const float* ew2  = (const float*)d_in[13];
  const float* ew3  = (const float*)d_in[14];
  float* out = (float*)d_out;

  char* ws = (char*)d_ws;
  size_t off = 0;
  auto alloc = [&](size_t bytes) -> char* {
    char* p = ws + off;
    off = (off + bytes + 255) & ~(size_t)255;
    return p;
  };
  const size_t TD = (size_t)Td * Dd, TFF = (size_t)Td * FFd;
  bf16* xbf  = (bf16*)alloc(TD * 2);
  bf16* wqT  = (bf16*)alloc((size_t)Dd * Dd * 2);
  bf16* wkT  = (bf16*)alloc((size_t)Dd * Dd * 2);
  bf16* wvT  = (bf16*)alloc((size_t)Dd * Dd * 2);
  bf16* woT  = (bf16*)alloc((size_t)Dd * Dd * 2);
  bf16* w1T  = (bf16*)alloc((size_t)FFd * Dd * 2);
  bf16* w3T  = (bf16*)alloc((size_t)FFd * Dd * 2);
  bf16* w2T  = (bf16*)alloc((size_t)Dd * FFd * 2);
  bf16* e1T  = (bf16*)alloc((size_t)Ed * EHd * Dd * 2);
  bf16* e3T  = (bf16*)alloc((size_t)Ed * EHd * Dd * 2);
  bf16* e2T  = (bf16*)alloc((size_t)Ed * Dd * EHd * 2);
  bf16* qbf  = (bf16*)alloc(TD * 2);
  bf16* kbf  = (bf16*)alloc(TD * 2);
  bf16* vT   = (bf16*)alloc(TD * 2);
  bf16* obf  = (bf16*)alloc(TD * 2);
  bf16* hbf  = (bf16*)alloc(TD * 2);
  bf16* gbf  = (bf16*)alloc(TFF * 2);
  float* t0  = (float*)alloc(TD * 4);
  float* h   = (float*)alloc(TD * 4);
  float* a1  = (float*)alloc(TFF * 4);
  float* moe = (float*)alloc(TD * 4);
  float* gate = (float*)alloc((size_t)Td * Ed * 4);

  // 1) precision conversion (activations row-major, weights transposed)
  cast_bf16_kernel<<<2048, 256, 0, stream>>>(x, xbf, TD);
  tcast_bf16_kernel<<<2048, 256, 0, stream>>>(wq, wqT, Dd, Dd);
  tcast_bf16_kernel<<<2048, 256, 0, stream>>>(wk, wkT, Dd, Dd);
  tcast_bf16_kernel<<<2048, 256, 0, stream>>>(wv, wvT, Dd, Dd);
  tcast_bf16_kernel<<<2048, 256, 0, stream>>>(wo, woT, Dd, Dd);
  tcast_bf16_kernel<<<2048, 256, 0, stream>>>(w1, w1T, Dd, FFd);
  tcast_bf16_kernel<<<2048, 256, 0, stream>>>(w3, w3T, Dd, FFd);
  tcast_bf16_kernel<<<2048, 256, 0, stream>>>(w2, w2T, FFd, Dd);
  for (int e = 0; e < Ed; ++e) {
    tcast_bf16_kernel<<<2048, 256, 0, stream>>>(
        ew1 + (size_t)e * Dd * EHd, e1T + (size_t)e * EHd * Dd, Dd, EHd);
    tcast_bf16_kernel<<<2048, 256, 0, stream>>>(
        ew3 + (size_t)e * Dd * EHd, e3T + (size_t)e * EHd * Dd, Dd, EHd);
    tcast_bf16_kernel<<<2048, 256, 0, stream>>>(
        ew2 + (size_t)e * EHd * Dd, e2T + (size_t)e * Dd * EHd, EHd, Dd);
  }

  // 2) attention projections (V written transposed for P@V B-fragments)
  launch_gemm(xbf, wqT, Td, Dd, Dd, nullptr, 0, nullptr, 0, qbf, nullptr, 0, nullptr, stream);
  launch_gemm(xbf, wkT, Td, Dd, Dd, nullptr, 0, nullptr, 0, kbf, nullptr, 0, nullptr, stream);
  launch_gemm(xbf, wvT, Td, Dd, Dd, nullptr, 0, nullptr, 0, nullptr, vT, Td, nullptr, stream);

  // 3) flash attention
  {
    dim3 g(Sd / 128, Hh, Bd);
    attn_kernel<<<g, 256, 0, stream>>>(qbf, kbf, vT, obf);
  }

  // 4) output projection + reordered-norm residual: h = x + norm(o@wo)*anw
  launch_gemm(obf, woT, Td, Dd, Dd, t0, 0, nullptr, 0, nullptr, nullptr, 0, nullptr, stream);
  rmsnorm_add_kernel<<<Td, 256, 0, stream>>>(x, t0, anw, h, hbf);

  // 5) dense SwiGLU FFN: g = silu(h@w1)*(h@w3); ffn = g@w2
  launch_gemm(hbf, w1T, Td, FFd, Dd, a1, 0, nullptr, 0, nullptr, nullptr, 0, nullptr, stream);
  launch_gemm(hbf, w3T, Td, FFd, Dd, nullptr, 0, nullptr, 0, gbf, nullptr, 0, a1, stream);
  launch_gemm(gbf, w2T, Td, Dd, FFd, t0, 0, nullptr, 0, nullptr, nullptr, 0, nullptr, stream);

  // 6) MoE: router gates then dropless experts accumulated with gate scale
  hipMemsetAsync(moe, 0, TD * 4, stream);
  router_kernel<<<Td / 8, 256, 0, stream>>>(x, rw, gate);
  for (int e = 0; e < Ed; ++e) {
    const bf16* w1e = e1T + (size_t)e * EHd * Dd;
    const bf16* w3e = e3T + (size_t)e * EHd * Dd;
    const bf16* w2e = e2T + (size_t)e * Dd * EHd;
    launch_gemm(xbf, w1e, Td, EHd, Dd, a1, 0, nullptr, 0, nullptr, nullptr, 0, nullptr, stream);
    launch_gemm(xbf, w3e, Td, EHd, Dd, nullptr, 0, nullptr, 0, gbf, nullptr, 0, a1, stream);
    launch_gemm(gbf, w2e, Td, Dd, EHd, moe, 1, gate + e, Ed, nullptr, nullptr, 0, nullptr, stream);
  }

  // 7) out = h + rmsnorm(ffn)*ffw + rmsnorm(moe)*mow
  final_kernel<<<Td, 256, 0, stream>>>(h, t0, ffw, moe, mow, out);
}
